// Attention_66640712565145
// MI455X (gfx1250) — matrix operand order repaired and checked by reading, not compiled, not run on hardware
//
#include <hip/hip_runtime.h>

typedef __bf16 bf16;
typedef __attribute__((ext_vector_type(4)))  __bf16 v4bf;
typedef __attribute__((ext_vector_type(8)))  __bf16 v8bf;
typedef __attribute__((ext_vector_type(16))) __bf16 v16bf;
typedef __attribute__((ext_vector_type(8)))  float  v8f;
typedef unsigned int v4u __attribute__((ext_vector_type(4)));
typedef int v4i __attribute__((ext_vector_type(4)));
typedef int v8i __attribute__((ext_vector_type(8)));

#define HEADS 12
#define DIMH  64
#define SEQ   1024
#define BATCH 16
#define DIM   768
#define INNER 768
#define MTOT  (BATCH * SEQ)   // 16384
#define ATT_SCALE 0.125f

// ---------------- WMMA helpers (CDNA5 bf16 16x16x32) ----------------

__device__ __forceinline__ v8f wmma_bf16(v16bf a, v16bf b, v8f c) {
  return __builtin_amdgcn_wmma_f32_16x16x32_bf16(
      false, a, false, b, (short)0, c, false, false);
}

// A-matrix 16x32 bf16 fragment from a row-major [16][ld] buffer.
// lanes 0-15 -> M=lane, K = 0..7 & 16..23 ; lanes 16-31 -> K = 8..15 & 24..31
__device__ __forceinline__ v16bf load_a_frag(const bf16* base, int ld) {
  int lane = threadIdx.x & 31;
  int half = lane >> 4, m = lane & 15;
  const bf16* p = base + m * ld + half * 8;
  union { v16bf v; v8bf h[2]; } u;
  u.h[0] = *(const v8bf*)p;
  u.h[1] = *(const v8bf*)(p + 16);
  return u.v;
}

// B-matrix 32x16 bf16 fragment from a K-transposed [16][ldk] buffer (bt[n][k]).
// lanes 0-15 -> N=lane, K = 0..7 & 16..23 ; lanes 16-31 -> K = 8..15 & 24..31
__device__ __forceinline__ v16bf load_b_frag(const bf16* bt, int ldk) {
  int lane = threadIdx.x & 31;
  int half = lane >> 4, n = lane & 15;
  const bf16* p = bt + n * ldk + half * 8;
  union { v16bf v; v8bf h[2]; } u;
  u.h[0] = *(const v8bf*)p;
  u.h[1] = *(const v8bf*)(p + 16);
  return u.v;
}

// ---------------- TDM: async tile DMA (global -> padded LDS) ----------------
// Tile: 64 bf16 per row (contiguous) x `rows`; row stride in memory = pitch_elems.
// pad_enable=1, pad_interval=32 dwords (code 4), pad_amount=4 dwords (code 3)
// => LDS row pitch = 64 + 8 = 72 bf16, matching the ld=72 fragment loaders.
__device__ __forceinline__ void tdm_load_tile64(unsigned lds_off, const bf16* gptr,
                                                unsigned rows, unsigned pitch_elems) {
  unsigned long long ga = (unsigned long long)(size_t)gptr;
  v4u g0 = { 1u,                                  // count = 1 valid descriptor
             lds_off,                             // lds_addr
             (unsigned)ga,                        // global_addr[31:0]
             (unsigned)((ga >> 32) & 0x01FFFFFFu) | 0x80000000u };  // ga[56:32] | type=2
  v8i g1 = { (int)0x07110000u,       // data_size=2B<<16 | pad_en<<20 | intv=4<<22 | amt=3<<25
             (int)(64u << 16),       // tensor_dim0 = 64
             (int)(rows << 16),      // tensor_dim0 hi=0 | tensor_dim1 = rows
             (int)(64u << 16),       // tensor_dim1 hi=0 | tile_dim0 = 64
             (int)rows,              // tile_dim1 = rows, tile_dim2 = 0
             (int)pitch_elems,       // tensor_dim0_stride
             0, 0 };
  v4i z4 = {0, 0, 0, 0};
#if __clang_major__ >= 23
  v8i z8 = {0, 0, 0, 0, 0, 0, 0, 0};
  __builtin_amdgcn_tensor_load_to_lds(g0, g1, z4, z4, z8, 0);
#else
  __builtin_amdgcn_tensor_load_to_lds(g0, g1, z4, z4, 0);
#endif
}

// ---------------- fp32 -> bf16 conversion ----------------

__global__ __launch_bounds__(256)
void cvt_f32_bf16(const float* __restrict__ s, bf16* __restrict__ d, int n4) {
  int i = blockIdx.x * blockDim.x + threadIdx.x;
  int stride = gridDim.x * blockDim.x;
  for (; i < n4; i += stride) {
    float4 f = ((const float4*)s)[i];
    v4bf o = { (bf16)f.x, (bf16)f.y, (bf16)f.z, (bf16)f.w };
    ((v4bf*)d)[i] = o;
  }
}

// ------- double-buffered bf16 GEMM body: 64x128 C tile, K-step 64 -------
// A tile staged by TDM DMA (wave 0); B tile software-pipelined via registers.
// As: 2 x [64][72] ; Bt: 2 x [128][72] (n-major, K-transposed)

template <int LDW>
__device__ __forceinline__ void gemm_tile(const bf16* __restrict__ X,
                                          const bf16* __restrict__ W,
                                          bf16* As, bf16* Bt,
                                          int mbase, int nbase, int Ktot,
                                          v8f acc[4]) {
  const int tid = threadIdx.x;
  const int wid = tid >> 5;
  const int wm = wid & 3;     // 4 wave-rows of 16
  const int wn = wid >> 2;    // 2 wave-cols of 64
  const int bkk = tid >> 4, bn0 = (tid & 15) * 8;  // B staging: 4 chunks (kk + 16*i)

  v8bf bReg[4];
  // prologue: DMA A tile 0, pipeline B tile 0 through registers
  if (wid == 0)
    tdm_load_tile64((unsigned)(size_t)As, &X[(size_t)mbase * Ktot], 64, (unsigned)Ktot);
#pragma unroll
  for (int i = 0; i < 4; ++i)
    bReg[i] = *(const v8bf*)&W[(bkk + 16 * i) * LDW + nbase + bn0];
#pragma unroll
  for (int i = 0; i < 4; ++i) {
    int kk = bkk + 16 * i;
#pragma unroll
    for (int j = 0; j < 8; ++j) Bt[(bn0 + j) * 72 + kk] = bReg[i][j];
  }
  if (wid == 0) __builtin_amdgcn_s_wait_tensorcnt(0);

  int cur = 0;
  for (int kt = 0; kt < Ktot; kt += 64) {
    __syncthreads();   // buffers[cur] ready; everyone done reading buffers[cur^1]
    const int nxt = cur ^ 1;
    const bool hasNext = (kt + 64) < Ktot;
    const bf16* Asc = As + cur * (64 * 72);
    const bf16* Btc = Bt + cur * (128 * 72);
    if (hasNext) {
      if (wid == 0)  // async DMA of next A tile into the other buffer
        tdm_load_tile64((unsigned)(size_t)(As + nxt * (64 * 72)),
                        &X[(size_t)mbase * Ktot + kt + 64], 64, (unsigned)Ktot);
#pragma unroll
      for (int i = 0; i < 4; ++i)  // issue next B loads; land under the WMMAs
        bReg[i] = *(const v8bf*)&W[(kt + 64 + bkk + 16 * i) * LDW + nbase + bn0];
    }
#pragma unroll
    for (int kk = 0; kk < 64; kk += 32) {
      v16bf a  = load_a_frag(&Asc[(wm * 16) * 72 + kk], 72);
      v16bf b0 = load_b_frag(&Btc[(wn * 64 +  0) * 72 + kk], 72);
      v16bf b1 = load_b_frag(&Btc[(wn * 64 + 16) * 72 + kk], 72);
      v16bf b2 = load_b_frag(&Btc[(wn * 64 + 32) * 72 + kk], 72);
      v16bf b3 = load_b_frag(&Btc[(wn * 64 + 48) * 72 + kk], 72);
      acc[0] = wmma_bf16(a, b0, acc[0]);
      acc[1] = wmma_bf16(a, b1, acc[1]);
      acc[2] = wmma_bf16(a, b2, acc[2]);
      acc[3] = wmma_bf16(a, b3, acc[3]);
    }
    if (hasNext) {
      bf16* Btn = Bt + nxt * (128 * 72);
#pragma unroll
      for (int i = 0; i < 4; ++i) {
        int kk = bkk + 16 * i;
#pragma unroll
        for (int j = 0; j < 8; ++j) Btn[(bn0 + j) * 72 + kk] = bReg[i][j];
      }
      if (wid == 0) __builtin_amdgcn_s_wait_tensorcnt(0);
    }
    cur = nxt;
  }
}

// ---------------- QV projection -> q,v in [b,h,n,d] bf16 ----------------

__global__ __launch_bounds__(256)
void gemm_qv(const bf16* __restrict__ X, const bf16* __restrict__ W,
             bf16* __restrict__ Q, bf16* __restrict__ V) {
  __shared__ bf16 As[2][64 * 72];
  __shared__ bf16 Bt[2][128 * 72];
  const int lane = threadIdx.x & 31;
  const int wid  = threadIdx.x >> 5;
  const int wm = wid & 3, wn = wid >> 2;
  const int mbase = blockIdx.x * 64, nbase = blockIdx.y * 128;

  v8f acc[4] = {v8f{}, v8f{}, v8f{}, v8f{}};
  gemm_tile<2 * INNER>(X, W, &As[0][0], &Bt[0][0], mbase, nbase, DIM, acc);

  int half = lane >> 4, nc = lane & 15;
#pragma unroll
  for (int nf = 0; nf < 4; ++nf) {
    int col = nbase + wn * 64 + nf * 16 + nc;
    bf16* dst = Q;
    int c2 = col;
    if (col >= INNER) { dst = V; c2 = col - INNER; }
    int h = c2 >> 6, d = c2 & 63;
#pragma unroll
    for (int r = 0; r < 8; ++r) {
      int m = mbase + wm * 16 + (half ? 8 + r : r);
      int b_ = m >> 10, n_ = m & 1023;
      dst[(((b_ * HEADS + h) << 10) + n_) * DIMH + d] = (bf16)acc[nf][r];
    }
  }
}

// ---------------- Output projection + bias -> fp32 ----------------

__global__ __launch_bounds__(256)
void gemm_out(const bf16* __restrict__ A, const bf16* __restrict__ W,
              const float* __restrict__ BIAS, float* __restrict__ OUT) {
  __shared__ bf16 As[2][64 * 72];
  __shared__ bf16 Bt[2][128 * 72];
  const int lane = threadIdx.x & 31;
  const int wid  = threadIdx.x >> 5;
  const int wm = wid & 3, wn = wid >> 2;
  const int mbase = blockIdx.x * 64, nbase = blockIdx.y * 128;

  v8f acc[4] = {v8f{}, v8f{}, v8f{}, v8f{}};
  gemm_tile<DIM>(A, W, &As[0][0], &Bt[0][0], mbase, nbase, INNER, acc);

  int half = lane >> 4, nc = lane & 15;
#pragma unroll
  for (int nf = 0; nf < 4; ++nf) {
    int col = nbase + wn * 64 + nf * 16 + nc;
    float bv = BIAS[col];
#pragma unroll
    for (int r = 0; r < 8; ++r) {
      int m = mbase + wm * 16 + (half ? 8 + r : r);
      OUT[(size_t)m * DIM + col] = acc[nf][r] + bv;
    }
  }
}

// -------- Flash attention: block = (bh, 64 query rows), 4 waves, ping-pong tiles ----

__global__ __launch_bounds__(128)
void attn_kernel(const bf16* __restrict__ Q, const bf16* __restrict__ K,
                 const bf16* __restrict__ V, const float* __restrict__ BIAS,
                 bf16* __restrict__ AO) {
  __shared__ bf16 k_lds[2][32 * 72];    // [key][d] ld=72 (TDM pad pitch)
  __shared__ bf16 vt_lds[2][64 * 40];   // [d][key] ld=40
  __shared__ bf16 p_lds[4 * 16 * 40];   // per-wave P patch

  const int tid = threadIdx.x, wid = tid >> 5, lane = tid & 31;
  const int bh = blockIdx.y;
  const int b  = bh / HEADS, h = bh % HEADS;
  const int qbase = blockIdx.x * 64 + wid * 16;
  const int half = lane >> 4, nc = lane & 15;
  const int vkey = tid >> 3, vd0 = (tid & 7) * 8;  // V staging geometry

  // prologue: DMA K tile 0, stage V tile 0, load q fragments
  if (wid == 0)
    tdm_load_tile64((unsigned)(size_t)&k_lds[0][0],
                    K + (size_t)h * SEQ * DIMH, 32, DIMH);
  const bf16* qptr = Q + ((size_t)bh * SEQ + qbase) * DIMH;
  v16bf qf0 = load_a_frag(qptr, DIMH);
  v16bf qf1 = load_a_frag(qptr + 32, DIMH);
  {
    const bf16* vg = V + (size_t)bh * SEQ * DIMH;
    v8bf vv0 = *(const v8bf*)&vg[vkey * DIMH + vd0];
    v8bf vv1 = *(const v8bf*)&vg[(vkey + 16) * DIMH + vd0];
#pragma unroll
    for (int j = 0; j < 8; ++j) {
      vt_lds[0][(vd0 + j) * 40 + vkey] = vv0[j];
      vt_lds[0][(vd0 + j) * 40 + vkey + 16] = vv1[j];
    }
  }
  if (wid == 0) __builtin_amdgcn_s_wait_tensorcnt(0);

  v8f o[4] = {v8f{}, v8f{}, v8f{}, v8f{}};
  float m_run[8], l_run[8];
#pragma unroll
  for (int r = 0; r < 8; ++r) { m_run[r] = -1e30f; l_run[r] = 0.f; }

  int cur = 0;
  for (int kt = 0; kt < SEQ; kt += 32) {
    __syncthreads();  // tiles[cur] ready; all waves done reading tiles[cur^1]
    const int nxt = cur ^ 1;
    const bool hasNext = (kt + 32) < SEQ;
    v8bf vv0, vv1;
    if (hasNext) {
      if (wid == 0)  // async DMA of next K tile into the other buffer
        tdm_load_tile64((unsigned)(size_t)&k_lds[nxt][0],
                        K + ((size_t)h * SEQ + kt + 32) * DIMH, 32, DIMH);
      const bf16* vg = V + ((size_t)bh * SEQ + kt + 32) * DIMH;
      vv0 = *(const v8bf*)&vg[vkey * DIMH + vd0];
      vv1 = *(const v8bf*)&vg[(vkey + 16) * DIMH + vd0];
      __builtin_prefetch(&BIAS[((size_t)h * SEQ + qbase + nc) * SEQ + kt + 32], 0, 0);
    }

    // S = q @ k^T : preload all 4 B fragments, then WMMA chain
    const bf16* kc = &k_lds[cur][0];
    v16bf kb0 = load_b_frag(&kc[0], 72);
    v16bf kb1 = load_b_frag(&kc[32], 72);
    v16bf kb2 = load_b_frag(&kc[16 * 72], 72);
    v16bf kb3 = load_b_frag(&kc[16 * 72 + 32], 72);
    v8f s0 = {}, s1 = {};
    s0 = wmma_bf16(qf0, kb0, s0);
    s0 = wmma_bf16(qf1, kb1, s0);
    s1 = wmma_bf16(qf0, kb2, s1);
    s1 = wmma_bf16(qf1, kb3, s1);

    // bias + scale + online softmax (fp32)
#pragma unroll
    for (int r = 0; r < 8; ++r) {
      int qrow = qbase + (half ? 8 + r : r);
      const float* bpr = BIAS + ((size_t)h * SEQ + qrow) * SEQ + kt;
      float x0 = (s0[r] + bpr[nc]) * ATT_SCALE;
      float x1 = (s1[r] + bpr[16 + nc]) * ATT_SCALE;
      float t = fmaxf(x0, x1);
#pragma unroll
      for (int off = 1; off < 16; off <<= 1)
        t = fmaxf(t, __shfl_xor(t, off));
      float mnew = fmaxf(m_run[r], t);
      float fr = __expf(m_run[r] - mnew);
      float p0 = __expf(x0 - mnew);
      float p1 = __expf(x1 - mnew);
      float rs = p0 + p1;
#pragma unroll
      for (int off = 1; off < 16; off <<= 1)
        rs += __shfl_xor(rs, off);
      l_run[r] = l_run[r] * fr + rs;
      m_run[r] = mnew;
      o[0][r] *= fr; o[1][r] *= fr; o[2][r] *= fr; o[3][r] *= fr;
      int rl = half ? 8 + r : r;
      p_lds[(wid * 16 + rl) * 40 + nc] = (bf16)p0;
      p_lds[(wid * 16 + rl) * 40 + 16 + nc] = (bf16)p1;
    }
    asm volatile("s_wait_dscnt 0" ::: "memory");  // own-wave P RAW fence

    const bf16* vc = &vt_lds[cur][0];
    v16bf pa  = load_a_frag(&p_lds[wid * 16 * 40], 40);
    v16bf bv0 = load_b_frag(&vc[0], 40);
    v16bf bv1 = load_b_frag(&vc[16 * 40], 40);
    v16bf bv2 = load_b_frag(&vc[32 * 40], 40);
    v16bf bv3 = load_b_frag(&vc[48 * 40], 40);
    o[0] = wmma_bf16(pa, bv0, o[0]);
    o[1] = wmma_bf16(pa, bv1, o[1]);
    o[2] = wmma_bf16(pa, bv2, o[2]);
    o[3] = wmma_bf16(pa, bv3, o[3]);

    if (hasNext) {  // scatter next V tile into the other buffer; wait DMA
#pragma unroll
      for (int j = 0; j < 8; ++j) {
        vt_lds[nxt][(vd0 + j) * 40 + vkey] = vv0[j];
        vt_lds[nxt][(vd0 + j) * 40 + vkey + 16] = vv1[j];
      }
      if (wid == 0) __builtin_amdgcn_s_wait_tensorcnt(0);
    }
    cur = nxt;
  }

  // normalize and write [b, n, h*64 + d] bf16
#pragma unroll
  for (int r = 0; r < 8; ++r) {
    float inv = 1.0f / l_run[r];
    int qrow = qbase + (half ? 8 + r : r);
    size_t orow = ((size_t)b * SEQ + qrow) * INNER + h * DIMH;
#pragma unroll
    for (int nf = 0; nf < 4; ++nf)
      AO[orow + nf * 16 + nc] = (bf16)(o[nf][r] * inv);
  }
}

// ---------------- launch ----------------

extern "C" void kernel_launch(void* const* d_in, const int* in_sizes, int n_in,
                              void* d_out, int out_size, void* d_ws, size_t ws_size,
                              hipStream_t stream) {
  (void)in_sizes; (void)n_in; (void)out_size; (void)ws_size;
  const float* x        = (const float*)d_in[0];
  const float* w_qv     = (const float*)d_in[1];
  const float* ext_k    = (const float*)d_in[2];
  const float* ext_bias = (const float*)d_in[3];
  const float* w_out    = (const float*)d_in[4];
  const float* b_out    = (const float*)d_in[5];
  float* out = (float*)d_out;

  char* ws = (char*)d_ws;
  size_t off = 0;
  auto carve = [&](size_t bytes) {
    void* p = ws + off;
    off = (off + bytes + 255) & ~(size_t)255;
    return p;
  };
  const size_t nX  = (size_t)MTOT * DIM;
  const size_t nWQ = (size_t)DIM * 2 * INNER;
  const size_t nK  = (size_t)HEADS * SEQ * DIMH;
  const size_t nWO = (size_t)INNER * DIM;
  const size_t nQV = (size_t)BATCH * HEADS * SEQ * DIMH;

  bf16* xb    = (bf16*)carve(nX * 2);
  bf16* wqvb  = (bf16*)carve(nWQ * 2);
  bf16* kb    = (bf16*)carve(nK * 2);
  bf16* woutb = (bf16*)carve(nWO * 2);
  bf16* qb    = (bf16*)carve(nQV * 2);
  bf16* vb    = (bf16*)carve(nQV * 2);
  bf16* aob   = (bf16*)carve(nX * 2);

  auto grid4 = [](size_t n4) {
    size_t g = (n4 + 255) / 256;
    return (unsigned)(g > 8192 ? 8192 : g);
  };
  cvt_f32_bf16<<<grid4(nX / 4),  256, 0, stream>>>(x,     xb,    (int)(nX / 4));
  cvt_f32_bf16<<<grid4(nWQ / 4), 256, 0, stream>>>(w_qv,  wqvb,  (int)(nWQ / 4));
  cvt_f32_bf16<<<grid4(nK / 4),  256, 0, stream>>>(ext_k, kb,    (int)(nK / 4));
  cvt_f32_bf16<<<grid4(nWO / 4), 256, 0, stream>>>(w_out, woutb, (int)(nWO / 4));

  gemm_qv<<<dim3(MTOT / 64, (2 * INNER) / 128), 256, 0, stream>>>(xb, wqvb, qb, vb);
  attn_kernel<<<dim3(SEQ / 64, BATCH * HEADS), 128, 0, stream>>>(qb, kb, vb, ext_bias, aob);
  gemm_out<<<dim3(MTOT / 64, DIM / 128), 256, 0, stream>>>(aob, woutb, b_out, out);
}
